// MultiHeadAttention_90829968376408
// MI455X (gfx1250) — matrix operand order repaired and checked by reading, not compiled, not run on hardware
//
#include <hip/hip_runtime.h>
#include <stdint.h>

typedef __attribute__((ext_vector_type(16))) __bf16 v16bf;
typedef __attribute__((ext_vector_type(8)))  float  v8f;
typedef __attribute__((ext_vector_type(4)))  unsigned int u32x4;
typedef __attribute__((ext_vector_type(8)))  int i32x8;
typedef __attribute__((ext_vector_type(4)))  int i32x4;

#define NSEQ   4096
#define DMODEL 1024
#define NHEADS 16
#define HDIM   64

#if __has_builtin(__builtin_amdgcn_tensor_load_to_lds)
#define HAVE_TDM 1
#else
#define HAVE_TDM 0
#endif

#if __has_builtin(__builtin_amdgcn_permlane16)
#define HAVE_PERMLANE 1
#else
#define HAVE_PERMLANE 0
#endif

__device__ __forceinline__ v8f wmma_bf16(v16bf a, v16bf b, v8f c) {
  return __builtin_amdgcn_wmma_f32_16x16x32_bf16(false, a, false, b, (short)0, c,
                                                 false, false);
}

// A-fragment (16x32 bf16, row-major source): lane L holds row (L&15);
// half = L>>4 selects K groups {half*8..half*8+7} and {16+half*8..}.
__device__ __forceinline__ v16bf frag_a(const __bf16* rowbase, int half) {
  union { v16bf v; uint32_t u[8]; } f;
  const uint32_t* p = (const uint32_t*)(rowbase + half * 8);
#pragma unroll
  for (int j = 0; j < 4; ++j) { f.u[j] = p[j]; f.u[4 + j] = p[8 + j]; }
  return f.v;
}

// B-fragment (32x16 bf16): lane L holds col (L&15); K groups {half*8..half*8+7}
// and {16+half*8..} when the source row is the "col" of B (B stored transposed).
__device__ __forceinline__ v16bf frag_b(const __bf16* rowbase, int half) {
  union { v16bf v; uint32_t u[8]; } f;
  const uint32_t* p = (const uint32_t*)(rowbase + half * 8);
#pragma unroll
  for (int j = 0; j < 4; ++j) { f.u[j] = p[j]; f.u[4 + j] = p[8 + j]; }
  return f.v;
}

// ---- 16-lane xor reductions: v_permlane16_b32 (VALU) instead of ds_bpermute.
#if HAVE_PERMLANE
__device__ __forceinline__ float xl16(float v, unsigned lo, unsigned hi) {
  union { float f; unsigned u; } a, r;
  a.f = v;
  r.u = __builtin_amdgcn_permlane16(a.u, a.u, lo, hi, false, false);
  return r.f;
}
__device__ __forceinline__ float rowmax16(float v) {
  v = fmaxf(v, xl16(v, 0x67452301u, 0xEFCDAB89u));  // xor 1
  v = fmaxf(v, xl16(v, 0x54761032u, 0xDCFE98BAu));  // xor 2
  v = fmaxf(v, xl16(v, 0x32107654u, 0xBA98FEDCu));  // xor 4
  v = fmaxf(v, xl16(v, 0xFEDCBA98u, 0x76543210u));  // xor 8
  return v;
}
__device__ __forceinline__ float rowsum16(float v) {
  v += xl16(v, 0x67452301u, 0xEFCDAB89u);
  v += xl16(v, 0x54761032u, 0xDCFE98BAu);
  v += xl16(v, 0x32107654u, 0xBA98FEDCu);
  v += xl16(v, 0xFEDCBA98u, 0x76543210u);
  return v;
}
#else
__device__ __forceinline__ float rowmax16(float v) {
#pragma unroll
  for (int m = 1; m < 16; m <<= 1) v = fmaxf(v, __shfl_xor(v, m, 32));
  return v;
}
__device__ __forceinline__ float rowsum16(float v) {
#pragma unroll
  for (int m = 1; m < 16; m <<= 1) v += __shfl_xor(v, m, 32);
  return v;
}
#endif

#if HAVE_TDM
// TDM: load a 2D bf16 tile [tile_d1 rows x tile_d0 elems], row stride
// t_stride elems in global, into LDS with pad_amount=1 dword every
// 2^(pad_iv+1) dwords (produces the +2-element padded LDS stride).
__device__ __forceinline__ void tdm_load_tile_2d(const __bf16* gsrc,
                                                 uint32_t lds_byte_off,
                                                 unsigned tile_d0, unsigned tile_d1,
                                                 unsigned t_stride, unsigned pad_iv) {
  const uint64_t ga = (uint64_t)(uintptr_t)gsrc;
  u32x4 g0;
  g0[0] = 1u;                                                // count=1
  g0[1] = lds_byte_off;                                      // lds_addr
  g0[2] = (uint32_t)ga;                                      // global_addr[31:0]
  g0[3] = (uint32_t)((ga >> 32) & 0x1FFFFFFu) | (2u << 30);  // addr[56:32]|type=2
  i32x8 g1;
  g1[0] = (int)((1u << 16) | (1u << 20) | (pad_iv << 22));   // 2B elems, pad_en
  g1[1] = (int)(tile_d0 << 16);                              // tensor_dim0[15:0]
  g1[2] = (int)((tile_d0 >> 16) | (tile_d1 << 16));          // dim0 hi | dim1 lo
  g1[3] = (int)((tile_d1 >> 16) | (tile_d0 << 16));          // dim1 hi | tile_dim0
  g1[4] = (int)(tile_d1 & 0xFFFFu);                          // tile_dim1
  g1[5] = (int)t_stride;                                     // tensor_dim0_stride
  g1[6] = 0;
  g1[7] = 0;
  __builtin_amdgcn_tensor_load_to_lds(g0, g1, (i32x4)0, (i32x4)0, (i32x8)0, 0);
}
#endif

// C[M x Nn] = A[M x K] @ B[K x Nn], C *= outScale.  Block tile 128x128,
// K-step 32, double-buffered staging; 8 wave32 arranged 2(M) x 4(N),
// wave tile 64x32.
template <bool A_BF16, bool OUT_BF16>
__global__ __launch_bounds__(256) void gemm128(const void* __restrict__ Araw,
                                               const float* __restrict__ B,
                                               void* __restrict__ Craw,
                                               int M, int K, int Nn,
                                               float outScale) {
  __shared__ __bf16 sA[2][128 * 34];   // [m][k], stride 34 (pad)
  __shared__ __bf16 sBT[2][128 * 34];  // [n][k], stride 34 (B transposed)

  const int tid = threadIdx.x;
  const int bm = blockIdx.x * 128;
  const int bn = blockIdx.y * 128;
  const int wid = tid >> 5, lane = tid & 31;
  const int wm = (wid & 1) * 64;
  const int wn = (wid >> 1) * 32;
  const int lm = lane & 15, half = lane >> 4;

  auto stage = [&](int k0, int buf) {
    if constexpr (A_BF16) {
#if HAVE_TDM
      if (wid == 0)  // 128x32 bf16 tile -> LDS stride 34 (pad 1dw / 16dw)
        tdm_load_tile_2d((const __bf16*)Araw + (size_t)bm * K + k0,
                         (uint32_t)(uintptr_t)&sA[buf][0], 32u, 128u, (unsigned)K, 3u);
#else
      const int row = tid >> 1;
      const int kp = (tid & 1) * 16;
      const uint32_t* s32 =
          (const uint32_t*)((const __bf16*)Araw + (size_t)(bm + row) * K + k0 + kp);
      uint32_t* d32 = (uint32_t*)&sA[buf][row * 34 + kp];
#pragma unroll
      for (int j = 0; j < 8; ++j) d32[j] = s32[j];
#endif
    } else {
      const int row = tid >> 1;
      const int kp = (tid & 1) * 16;
      const float* src = (const float*)Araw + (size_t)(bm + row) * K + k0 + kp;
      __bf16* dst = &sA[buf][row * 34 + kp];
#pragma unroll
      for (int j = 0; j < 16; ++j) dst[j] = (__bf16)src[j];
    }
    {  // B tile 32x128, stored transposed [n][k]
      const int col = tid >> 1;
      const int kp = (tid & 1) * 16;
      const float* src = B + (size_t)(k0 + kp) * Nn + bn + col;
      __bf16* dst = &sBT[buf][col * 34 + kp];
#pragma unroll
      for (int j = 0; j < 16; ++j) dst[j] = (__bf16)src[(size_t)j * Nn];
    }
  };

  v8f acc[4][2] = {};
  const int nk = K / 32;
  stage(0, 0);

  for (int i = 0; i < nk; ++i) {
#if HAVE_TDM
    if constexpr (A_BF16) __builtin_amdgcn_s_wait_tensorcnt(0);  // tile i done
#endif
    __syncthreads();  // retires buf (i-1) reads + all V/B ds stores of tile i
    if (i + 1 < nk) stage((i + 1) * 32, (i + 1) & 1);  // overlap with compute

    const __bf16* As = &sA[i & 1][0];
    const __bf16* Bs = &sBT[i & 1][0];
    v16bf af[4], bfr[2];
#pragma unroll
    for (int a = 0; a < 4; ++a) af[a] = frag_a(&As[(wm + 16 * a + lm) * 34], half);
#pragma unroll
    for (int b = 0; b < 2; ++b) bfr[b] = frag_b(&Bs[(wn + 16 * b + lm) * 34], half);
#pragma unroll
    for (int a = 0; a < 4; ++a)
#pragma unroll
      for (int b = 0; b < 2; ++b) acc[a][b] = wmma_bf16(af[a], bfr[b], acc[a][b]);
  }

#pragma unroll
  for (int i = 0; i < 4; ++i)
#pragma unroll
    for (int j = 0; j < 2; ++j)
#pragma unroll
      for (int r = 0; r < 8; ++r) {
        const int row = bm + wm + 16 * i + r + 8 * half;  // C: VGPR r -> M=r+8*half
        const int col = bn + wn + 16 * j + lm;
        const float val = acc[i][j][r] * outScale;
        if constexpr (OUT_BF16)
          ((__bf16*)Craw)[(size_t)row * Nn + col] = (__bf16)val;
        else
          ((float*)Craw)[(size_t)row * Nn + col] = val;
      }
}

// Flash attention: grid (NSEQ/128, NHEADS); 8 waves, 16 query rows per wave,
// 64-key tiles, double-buffered K (TDM) / V (manual transpose) staging.
// Q pre-scaled by 0.125*log2(e) so softmax runs in exp2 domain.
__global__ __launch_bounds__(256) void flash_attn(const __bf16* __restrict__ Qb,
                                                  const __bf16* __restrict__ Kb,
                                                  const __bf16* __restrict__ Vb,
                                                  __bf16* __restrict__ Hb) {
  __shared__ __bf16 sK[2][64 * 66];   // [key][dim], stride 66
  __shared__ __bf16 sVT[2][64 * 66];  // [dim][key], stride 66
  __shared__ __bf16 sP[8][16 * 64];   // per-wave P scratch, stride 64

  const int head = blockIdx.y;
  const int q0 = blockIdx.x * 128;
  const int tid = threadIdx.x, wid = tid >> 5, lane = tid & 31;
  const int lm = lane & 15, half = lane >> 4;
  const int qrow = q0 + wid * 16;

  auto stageKV = [&](int kb, int buf) {
#if HAVE_TDM
    if (wid == 0)  // 64x64 bf16 K tile -> LDS stride 66 (pad 1dw / 32dw)
      tdm_load_tile_2d(Kb + (size_t)kb * DMODEL + head * HDIM,
                       (uint32_t)(uintptr_t)&sK[buf][0], (unsigned)HDIM, 64u,
                       (unsigned)DMODEL, 4u);
#else
    {
      const int key = tid >> 2;
      const int dp = (tid & 3) * 16;
      const uint32_t* s32 =
          (const uint32_t*)(Kb + (size_t)(kb + key) * DMODEL + head * HDIM + dp);
      uint32_t* d32 = (uint32_t*)&sK[buf][key * 66 + dp];
#pragma unroll
      for (int j = 0; j < 8; ++j) d32[j] = s32[j];
    }
#endif
    {  // V tile 64x64, stored transposed [dim][key]
      const int key = tid >> 2;
      const int dp = (tid & 3) * 16;
      const __bf16* vsrc = Vb + (size_t)(kb + key) * DMODEL + head * HDIM + dp;
#pragma unroll
      for (int j = 0; j < 16; ++j) sVT[buf][(dp + j) * 66 + key] = vsrc[j];
    }
  };

  v16bf qf[2];
  {
    const __bf16* qp = Qb + (size_t)(qrow + lm) * DMODEL + head * HDIM;
#pragma unroll
    for (int ks = 0; ks < 2; ++ks) qf[ks] = frag_a(qp + ks * 32, half);
  }

  v8f o[4] = {};
  float mrow[8], lrow[8];
#pragma unroll
  for (int r = 0; r < 8; ++r) { mrow[r] = -3.0e38f; lrow[r] = 0.f; }

  const int nt = NSEQ / 64;
  stageKV(0, 0);

  for (int i = 0; i < nt; ++i) {
#if HAVE_TDM
    __builtin_amdgcn_s_wait_tensorcnt(0);  // K tile i landed in LDS
#endif
    __syncthreads();  // retires buf (i-1) reads + tile i V stores
    if (i + 1 < nt) stageKV((i + 1) * 64, (i + 1) & 1);  // overlap with compute

    const __bf16* kbuf = &sK[i & 1][0];
    const __bf16* vbuf = &sVT[i & 1][0];

    // S(16x64) = Qs @ K^T : row-major K rows feed B-frags directly.
    v8f s[4] = {};
#pragma unroll
    for (int ks = 0; ks < 2; ++ks)
#pragma unroll
      for (int n = 0; n < 4; ++n) {
        v16bf bk = frag_b(&kbuf[(n * 16 + lm) * 66 + ks * 32], half);
        s[n] = wmma_bf16(qf[ks], bk, s[n]);
      }

    // Online softmax (exp2 domain; scale folded into Q).
    float e[4][8];
#pragma unroll
    for (int r = 0; r < 8; ++r) {
      float t = fmaxf(fmaxf(s[0][r], s[1][r]), fmaxf(s[2][r], s[3][r]));
      t = rowmax16(t);
      const float mn = fmaxf(mrow[r], t);
      const float alpha = exp2f(mrow[r] - mn);
      float rs = 0.f;
#pragma unroll
      for (int n = 0; n < 4; ++n) { e[n][r] = exp2f(s[n][r] - mn); rs += e[n][r]; }
      rs = rowsum16(rs);
      lrow[r] = lrow[r] * alpha + rs;
      mrow[r] = mn;
#pragma unroll
      for (int f = 0; f < 4; ++f) o[f][r] = o[f][r] * alpha;
    }

    // C-layout P -> A-layout via per-wave LDS buffer.
    __bf16* pw = &sP[wid][0];
#pragma unroll
    for (int r = 0; r < 8; ++r) {
      const int prow = r + 8 * half;
#pragma unroll
      for (int n = 0; n < 4; ++n) pw[prow * 64 + n * 16 + lm] = (__bf16)e[n][r];
    }
    asm volatile("s_wait_dscnt 0" ::: "memory");  // wave-internal LDS RAW

#pragma unroll
    for (int kk = 0; kk < 2; ++kk) {
      v16bf pf = frag_a(&pw[lm * 64 + kk * 32], half);
#pragma unroll
      for (int f = 0; f < 4; ++f) {
        v16bf bv = frag_b(&vbuf[(f * 16 + lm) * 66 + kk * 32], half);
        o[f] = wmma_bf16(pf, bv, o[f]);
      }
    }
  }

#pragma unroll
  for (int r = 0; r < 8; ++r) {
    const float inv = 1.0f / lrow[r];
    const int row = qrow + r + 8 * half;
#pragma unroll
    for (int f = 0; f < 4; ++f)
      Hb[(size_t)row * DMODEL + head * HDIM + f * 16 + lm] = (__bf16)(o[f][r] * inv);
  }
}

extern "C" void kernel_launch(void* const* d_in, const int* in_sizes, int n_in,
                              void* d_out, int out_size, void* d_ws, size_t ws_size,
                              hipStream_t stream) {
  const float* q  = (const float*)d_in[0];
  const float* k  = (const float*)d_in[1];
  const float* v  = (const float*)d_in[2];
  const float* Wq = (const float*)d_in[3];
  const float* Wk = (const float*)d_in[4];
  const float* Wv = (const float*)d_in[5];
  const float* Wo = (const float*)d_in[6];

  const size_t mat = (size_t)NSEQ * DMODEL;
  __bf16* Qb = (__bf16*)d_ws;  // 8 MB each, 32 MB total
  __bf16* Kb = Qb + mat;
  __bf16* Vb = Kb + mat;
  __bf16* Hb = Vb + mat;

  dim3 blk(256);
  dim3 gg(NSEQ / 128, DMODEL / 128);

  // softmax scale (1/sqrt(64)) and log2(e) folded into the Q projection
  const float qScale = 0.125f * 1.44269504088896f;

  gemm128<false, true><<<gg, blk, 0, stream>>>(q, Wq, Qb, NSEQ, DMODEL, DMODEL, qScale);
  gemm128<false, true><<<gg, blk, 0, stream>>>(k, Wk, Kb, NSEQ, DMODEL, DMODEL, 1.0f);
  gemm128<false, true><<<gg, blk, 0, stream>>>(v, Wv, Vb, NSEQ, DMODEL, DMODEL, 1.0f);
  flash_attn<<<dim3(NSEQ / 128, NHEADS), blk, 0, stream>>>(Qb, Kb, Vb, Hb);
  gemm128<true, false><<<gg, blk, 0, stream>>>(Hb, Wo, (float*)d_out, NSEQ, DMODEL,
                                               DMODEL, 1.0f);
}
